// LSTM_model_21509196218925
// MI455X (gfx1250) — compile-verified, matrix operand order checked by reading, not run on hardware
//
#include <hip/hip_runtime.h>

// Problem dims
#define T_STEPS 512
#define BATCH   512
#define SDIM    64
#define HID     128
#define GATES   512          // 4*HID, gate order (i,f,g,o)
#define ADIM    10
// Tiling
#define BT      16           // batch rows per block (one WMMA M-tile)
#define NBLK    (BATCH / BT) // 32 blocks, fully independent (no cross-WG sync)
#define NW      16           // waves per block; wave w owns gate cols [32w, 32w+32)
#define NTHREADS (NW * 32)
#define GSTRIDE 516          // padded gate-matrix stride (breaks LDS bank conflicts)
#define W1STRIDE 136         // padded Whh1 LDS row stride (bf16 elems; 272B, 16B-aligned)

typedef __attribute__((ext_vector_type(16))) __bf16 v16bf;
typedef __attribute__((ext_vector_type(8)))  float  v8f;

// ---- WMMA fragment builders -------------------------------------------------
// 16-bit A/B operand layout (CDNA5 ISA 7.12.2): lanes 0-15 hold K {base+0..7,
// base+16..23}, lanes 16-31 hold K {base+8..15, base+24..31}; VGPR i = pair
// (2i, 2i+1) -> two contiguous 8-element (16B) runs per lane.

__device__ __forceinline__ v16bf frag_from_f32(const float* __restrict__ p) {
  v16bf r;
#pragma unroll
  for (int i = 0; i < 8; ++i) r[i] = (__bf16)p[i];
#pragma unroll
  for (int i = 0; i < 8; ++i) r[8 + i] = (__bf16)p[16 + i];
  return r;
}

__device__ __forceinline__ v16bf frag_from_lds(const __bf16* p) {
  // two contiguous 16B LDS reads -> ds_load_b128 x2
  v16bf r;
#pragma unroll
  for (int i = 0; i < 8; ++i) r[i] = p[i];
#pragma unroll
  for (int i = 0; i < 8; ++i) r[8 + i] = p[16 + i];
  return r;
}

#define WMMA_BF16(acc, a, b) \
  __builtin_amdgcn_wmma_f32_16x16x32_bf16(false, (a), false, (b), (short)0, (acc), false, false)

__device__ __forceinline__ float fast_sigmoid(float v) {
  return 1.0f / (1.0f + __expf(-v));
}
__device__ __forceinline__ float fast_tanh(float v) {
  return 1.0f - 2.0f / (__expf(2.0f * v) + 1.0f);   // saturates correctly at +-1
}

__global__ __launch_bounds__(NTHREADS, 1)
void lstm_persist(const float* __restrict__ x,
                  const float* __restrict__ Wih0, const float* __restrict__ Whh0,
                  const float* __restrict__ bih0, const float* __restrict__ bhh0,
                  const float* __restrict__ Wih1, const float* __restrict__ Whh1,
                  const float* __restrict__ bih1, const float* __restrict__ bhh1,
                  const float* __restrict__ Wout, const float* __restrict__ bout,
                  float* __restrict__ out) {
  __shared__ __bf16 s_x [BT * SDIM];       // staged x_t (bf16)
  __shared__ __bf16 s_h0[BT * HID];        // layer0 hidden state (GEMM A input)
  __shared__ __bf16 s_h1[BT * HID];        // layer1 hidden state
  __shared__ __bf16 s_w1h[GATES * W1STRIDE]; // Whh1 (bf16, padded rows) ~139KB
  __shared__ float  s_g [BT * GSTRIDE];    // gate pre-activations (fp32)
  __shared__ float  s_b0[GATES];           // b_ih0 + b_hh0
  __shared__ float  s_b1[GATES];
  __shared__ float  s_wo[ADIM * HID];
  __shared__ float  s_bo[ADIM];

  const int tid   = threadIdx.x;
  const int lane  = tid & 31;
  const int wv    = tid >> 5;          // wave 0..15
  const int brow0 = blockIdx.x * BT;   // first batch row of this block

  // ---- one-time LDS init -----------------------------------------------------
  for (int i = tid; i < BT * HID; i += NTHREADS) { s_h0[i] = (__bf16)0.0f; s_h1[i] = (__bf16)0.0f; }
  for (int i = tid; i < GATES; i += NTHREADS)    { s_b0[i] = bih0[i] + bhh0[i]; s_b1[i] = bih1[i] + bhh1[i]; }
  for (int i = tid; i < ADIM * HID; i += NTHREADS) s_wo[i] = Wout[i];
  if (tid < ADIM) s_bo[tid] = bout[tid];
  // Whh1 -> LDS bf16 with padded stride (read as B fragments each step)
  for (int i = tid; i < GATES * HID; i += NTHREADS) {
    const int r = i >> 7, c = i & (HID - 1);
    s_w1h[r * W1STRIDE + c] = (__bf16)Whh1[i];
  }

  // ---- one-time weight load into registers (weight-stationary) ---------------
  // gates = A @ W^T  =>  B[k][n] = W[n][k]; W rows are contiguous in K, so each
  // lane (column n = j) reads contiguous K runs of its own W row.
  const int koff = (lane & 16) ? 8 : 0;
  const int j0   = wv * 32 + (lane & 15);       // N-tile 0 column / W row
  const int j1   = j0 + 16;                     // N-tile 1

  v16bf w0[6][2];   // layer0: kt 0..1 from Wih0 (K=64), kt 2..5 from Whh0 (K=128)
#pragma unroll
  for (int nt = 0; nt < 2; ++nt) {
    const int j = nt ? j1 : j0;
#pragma unroll
    for (int kt = 0; kt < 2; ++kt)
      w0[kt][nt] = frag_from_f32(Wih0 + (size_t)j * SDIM + kt * 32 + koff);
#pragma unroll
    for (int kt = 2; kt < 6; ++kt)
      w0[kt][nt] = frag_from_f32(Whh0 + (size_t)j * HID + (kt - 2) * 32 + koff);
  }
  v16bf w1[4][2];   // layer1 h0-half only: kt 0..3 from Wih1 (K=128)
#pragma unroll
  for (int nt = 0; nt < 2; ++nt) {
    const int j = nt ? j1 : j0;
#pragma unroll
    for (int kt = 0; kt < 4; ++kt)
      w1[kt][nt] = frag_from_f32(Wih1 + (size_t)j * HID + kt * 32 + koff);
  }

  // persistent cell state: thread owns row um, hidden cols uj..uj+3
  float c0r[4] = {0.f, 0.f, 0.f, 0.f};
  float c1r[4] = {0.f, 0.f, 0.f, 0.f};
  const int um = tid >> 5;          // 0..15
  const int uj = (tid & 31) * 4;    // 0..124

  const int am   = lane & 15;               // A-fragment row (M)
  const int hi8  = (lane & 16) ? 8 : 0;     // D layout: lanes 16-31 hold M=8+r
  const int gcol = wv * 32 + (lane & 15);   // D column for N-tile 0

  __syncthreads();

  for (int t = 0; t < T_STEPS; ++t) {
    // -- stage x_t tile -> bf16 LDS (only global read in the loop) -------------
    {
      const float* xp = x + ((size_t)t * BATCH + brow0) * SDIM;
      for (int i = tid; i < BT * SDIM; i += NTHREADS) s_x[i] = (__bf16)xp[i];
    }
    __syncthreads();

    // -- layer0 GEMM: gates = [x_t | h0] @ [Wih0 | Whh0]^T ---------------------
    v8f acc0 = {0.f,0.f,0.f,0.f,0.f,0.f,0.f,0.f};
    v8f acc1 = {0.f,0.f,0.f,0.f,0.f,0.f,0.f,0.f};
    {
      // double-buffered A fragments: x (kt 0..1), then h0 (kt 2..5)
      v16bf aCur = frag_from_lds(s_x + am * SDIM + koff);
#pragma unroll
      for (int kt = 0; kt < 6; ++kt) {
        v16bf aNext;
        if (kt < 1)      aNext = frag_from_lds(s_x  + am * SDIM + (kt + 1) * 32 + koff);
        else if (kt < 5) aNext = frag_from_lds(s_h0 + am * HID + (kt - 1) * 32 + koff);
        acc0 = WMMA_BF16(acc0, aCur, w0[kt][0]);
        acc1 = WMMA_BF16(acc1, aCur, w0[kt][1]);
        if (kt < 5) aCur = aNext;
      }
    }
#pragma unroll
    for (int r = 0; r < 8; ++r) {          // D: VGPR r = row r (lo lanes) / 8+r (hi)
      s_g[(r + hi8) * GSTRIDE + gcol]      = acc0[r];
      s_g[(r + hi8) * GSTRIDE + gcol + 16] = acc1[r];
    }
    __syncthreads();

    // -- layer0 cell update ----------------------------------------------------
#pragma unroll
    for (int e = 0; e < 4; ++e) {
      const int j = uj + e;
      const float ig = s_g[um * GSTRIDE + j]       + s_b0[j];
      const float fg = s_g[um * GSTRIDE + j + 128] + s_b0[j + 128];
      const float gg = s_g[um * GSTRIDE + j + 256] + s_b0[j + 256];
      const float og = s_g[um * GSTRIDE + j + 384] + s_b0[j + 384];
      const float c  = fast_sigmoid(fg) * c0r[e] + fast_sigmoid(ig) * fast_tanh(gg);
      c0r[e] = c;
      s_h0[um * HID + j] = (__bf16)(fast_sigmoid(og) * fast_tanh(c));
    }
    __syncthreads();

    // -- layer1 GEMM: gates = [h0n | h1] @ [Wih1 | Whh1]^T ---------------------
    acc0 = (v8f){0.f,0.f,0.f,0.f,0.f,0.f,0.f,0.f};
    acc1 = (v8f){0.f,0.f,0.f,0.f,0.f,0.f,0.f,0.f};
    {
      // kt 0..3: A from h0 (new), B = Wih1 from registers
      v16bf aCur = frag_from_lds(s_h0 + am * HID + koff);
#pragma unroll
      for (int kt = 0; kt < 4; ++kt) {
        v16bf aNext;
        if (kt < 3) aNext = frag_from_lds(s_h0 + am * HID + (kt + 1) * 32 + koff);
        else        aNext = frag_from_lds(s_h1 + am * HID + koff);
        acc0 = WMMA_BF16(acc0, aCur, w1[kt][0]);
        acc1 = WMMA_BF16(acc1, aCur, w1[kt][1]);
        aCur = aNext;
      }
      // kt 4..7: A from h1 (prev step), B = Whh1 fragments read JIT from LDS
#pragma unroll
      for (int kt = 0; kt < 4; ++kt) {
        v16bf b0 = frag_from_lds(s_w1h + (size_t)j0 * W1STRIDE + kt * 32 + koff);
        v16bf b1 = frag_from_lds(s_w1h + (size_t)j1 * W1STRIDE + kt * 32 + koff);
        v16bf aNext;
        if (kt < 3) aNext = frag_from_lds(s_h1 + am * HID + (kt + 1) * 32 + koff);
        acc0 = WMMA_BF16(acc0, aCur, b0);
        acc1 = WMMA_BF16(acc1, aCur, b1);
        if (kt < 3) aCur = aNext;
      }
    }
#pragma unroll
    for (int r = 0; r < 8; ++r) {
      s_g[(r + hi8) * GSTRIDE + gcol]      = acc0[r];
      s_g[(r + hi8) * GSTRIDE + gcol + 16] = acc1[r];
    }
    __syncthreads();

    // -- layer1 cell update ----------------------------------------------------
#pragma unroll
    for (int e = 0; e < 4; ++e) {
      const int j = uj + e;
      const float ig = s_g[um * GSTRIDE + j]       + s_b1[j];
      const float fg = s_g[um * GSTRIDE + j + 128] + s_b1[j + 128];
      const float gg = s_g[um * GSTRIDE + j + 256] + s_b1[j + 256];
      const float og = s_g[um * GSTRIDE + j + 384] + s_b1[j + 384];
      const float c  = fast_sigmoid(fg) * c1r[e] + fast_sigmoid(ig) * fast_tanh(gg);
      c1r[e] = c;
      s_h1[um * HID + j] = (__bf16)(fast_sigmoid(og) * fast_tanh(c));
    }
    __syncthreads();

    // -- inline output projection: action = tanh(h1 @ Wout^T + bout) -----------
    if (tid < BT * ADIM) {                  // 160 threads = waves 0..4
      const int m = tid / ADIM;
      const int a = tid - m * ADIM;
      float s = s_bo[a];
#pragma unroll 8
      for (int j = 0; j < HID; ++j)
        s += (float)s_h1[m * HID + j] * s_wo[a * HID + j];
      out[((size_t)t * BATCH + brow0 + m) * ADIM + a] = fast_tanh(s);
    }
    // no barrier needed here: next write to s_h1 is 4 barriers away
  }
}

extern "C" void kernel_launch(void* const* d_in, const int* in_sizes, int n_in,
                              void* d_out, int out_size, void* d_ws, size_t ws_size,
                              hipStream_t stream) {
  (void)in_sizes; (void)n_in; (void)d_ws; (void)ws_size; (void)out_size;
  const float* x    = (const float*)d_in[0];
  const float* Wih0 = (const float*)d_in[1];
  const float* Whh0 = (const float*)d_in[2];
  const float* bih0 = (const float*)d_in[3];
  const float* bhh0 = (const float*)d_in[4];
  const float* Wih1 = (const float*)d_in[5];
  const float* Whh1 = (const float*)d_in[6];
  const float* bih1 = (const float*)d_in[7];
  const float* bhh1 = (const float*)d_in[8];
  const float* Wout = (const float*)d_in[9];
  const float* bout = (const float*)d_in[10];
  lstm_persist<<<dim3(NBLK), dim3(NTHREADS), 0, stream>>>(
      x, Wih0, Whh0, bih0, bhh0, Wih1, Whh1, bih1, bhh1, Wout, bout,
      (float*)d_out);
}
